// MonotonicAttention_73306501808938
// MI455X (gfx1250) — matrix operand order repaired and checked by reading, not compiled, not run on hardware
//
#include <hip/hip_runtime.h>
#include <hip/hip_bf16.h>

// ---- problem constants ----
#define TT 512
#define SS 512
#define BB 8
#define EE 1024
#define HH 16
#define DD 64
#define NN (BB * HH)          // 128 head-rows
#define NTS ((long)NN * TT * SS)
#define QKV ((long)TT * BB * EE)

typedef __attribute__((ext_vector_type(16))) _Float16 v16h;
typedef __attribute__((ext_vector_type(8)))  float    v8f;

// Epilogue modes
#define EPI_SCALE   0   // y = x * scale
#define EPI_SIGMOID 1   // y = sigmoid(x + aux[0])
#define EPI_BIAS    2   // y = x + aux[n]

union PackF16 { unsigned u; _Float16 h[2]; };

// ---------------------------------------------------------------------------
// Generic batched, strided WMMA GEMM:  C[m,n] = sum_k A[m,k] * B[k,n]
//   A columns contiguous (a_cs==1). C columns contiguous (c_cs==1).
//   batch offset = (z / bimod)*bo + (z % bimod)*bi   (covers linear + head split)
// Block: 256 threads = 8 waves; tile 128(M) x 64*WB(N); K step 32.
// Waves in 4x2 grid; each wave owns 32 x 32*WB  = 2 x 2*WB v_wmma accumulators.
// Double-buffered LDS, two-phase staging (batch ALL global loads first, then
// convert+packed stores) for memory-level parallelism, one barrier per K-step,
// global_prefetch on the tile-after-next.
// ---------------------------------------------------------------------------
template <int WB>
__global__ __launch_bounds__(256)
void gemm_wmma_f16(const float* __restrict__ A, long a_rs, long a_bo, long a_bi, int a_bim,
                   const float* __restrict__ Bm, long b_rs, long b_cs, long b_bo, long b_bi, int b_bim,
                   float* __restrict__ C, long c_rs, long c_bo, long c_bi, int c_bim,
                   int M, int Ncols, int K,
                   int mode, const float* __restrict__ aux, float scale)
{
    constexpr int NT = 64 * WB;                 // N tile
    __shared__ _Float16 As[2][128 * 32];        // [m][k]
    __shared__ _Float16 Bs[2][NT * 32];         // [n][k] (transposed: contiguous K per lane)

    const int z = blockIdx.z;
    const long aoff = (long)(z / a_bim) * a_bo + (long)(z % a_bim) * a_bi;
    const long boff = (long)(z / b_bim) * b_bo + (long)(z % b_bim) * b_bi;
    const long coff = (long)(z / c_bim) * c_bo + (long)(z % c_bim) * c_bi;

    const int bm0 = blockIdx.y * 128;
    const int bn0 = blockIdx.x * NT;

    const int tid  = threadIdx.x;
    const int lane = tid & 31;
    const int wid  = tid >> 5;                  // 0..7
    const int wm   = (wid >> 1) * 32;           // wave row offset in tile (4 rows of waves)
    const int wn   = (wid & 1) * (32 * WB);     // wave col offset in tile (2 cols of waves)
    const int half = lane >> 4;
    const int l15  = lane & 15;

    const float bias0 = (mode == EPI_SIGMOID) ? aux[0] : 0.0f;

    // Stage one 128x32 A tile + NTx32 B tile (f32 -> packed f16) into buffer.
    // Two phases: (1) issue ALL independent global loads into register arrays,
    // (2) convert + packed b32 LDS stores. Keeps many loads in flight.
    auto stage = [&](int kk, int buf) {
        unsigned* Ad = reinterpret_cast<unsigned*>(&As[buf][0]);
        unsigned* Bd = reinterpret_cast<unsigned*>(&Bs[buf][0]);

        float a0[8], a1[8];
        #pragma unroll
        for (int i = 0; i < 8; ++i) {           // A: 2048 f16-pairs, float2 loads
            int p = tid + i * 256;
            int r = p >> 4, c2 = p & 15;
            const float2 t = *reinterpret_cast<const float2*>(
                A + aoff + (long)(bm0 + r) * a_rs + (kk + 2 * c2));
            a0[i] = t.x; a1[i] = t.y;
        }
        float b0[4 * WB], b1[4 * WB];
        #pragma unroll
        for (int i = 0; i < 4 * WB; ++i) {      // B: NT*16 f16-pairs
            int p = tid + i * 256;
            int n = p >> 4, c2 = p & 15;
            const float* gb = Bm + boff + (long)(kk + 2 * c2) * b_rs + (long)(bn0 + n) * b_cs;
            b0[i] = gb[0];
            b1[i] = gb[b_rs];
        }
        // prefetch tile-after-next into cache hierarchy (global_prefetch_b8)
        if (kk + 32 < K) {
            __builtin_prefetch((const void*)(A + aoff + (long)(bm0 + (tid >> 1)) * a_rs + (kk + 32)), 0, 0);
            __builtin_prefetch((const void*)(Bm + boff + (long)(kk + 32 + (tid & 1) * 16) * b_rs
                                             + (long)(bn0 + (tid >> 1) * WB) * b_cs), 0, 0);
        }
        #pragma unroll
        for (int i = 0; i < 8; ++i) {
            int p = tid + i * 256;
            PackF16 pk; pk.h[0] = (_Float16)a0[i]; pk.h[1] = (_Float16)a1[i];
            Ad[p] = pk.u;                       // As[r][2*c2 .. 2*c2+1]
        }
        #pragma unroll
        for (int i = 0; i < 4 * WB; ++i) {
            int p = tid + i * 256;
            PackF16 pk; pk.h[0] = (_Float16)b0[i]; pk.h[1] = (_Float16)b1[i];
            Bd[p] = pk.u;                       // Bs[n][2*c2 .. 2*c2+1]
        }
    };

    v8f acc[2][2 * WB] = {};

    stage(0, 0);
    __syncthreads();

    const int ntiles = K >> 5;
    for (int kt = 0; kt < ntiles; ++kt) {
        const int cur = kt & 1;
        if (kt + 1 < ntiles) stage((kt + 1) << 5, cur ^ 1);   // fill other buffer

        // ---- build fragments (ISA 7.12.2 layouts) ----
        v16h af[2], bf[2 * WB];
        #pragma unroll
        for (int i = 0; i < 2; ++i) {
            int arow = wm + i * 16 + l15;
            #pragma unroll
            for (int e = 0; e < 16; ++e) {
                int kA = (e < 8) ? (half * 8 + e) : (16 + half * 8 + (e - 8));
                af[i][e] = As[cur][arow * 32 + kA];
            }
        }
        #pragma unroll
        for (int j = 0; j < 2 * WB; ++j) {
            int bcol = wn + j * 16 + l15;
            #pragma unroll
            for (int e = 0; e < 16; ++e) {
                int kB = (e < 8) ? (half * 8 + e) : (16 + half * 8 + (e - 8));
                bf[j][e] = Bs[cur][bcol * 32 + kB];
            }
        }
        #pragma unroll
        for (int i = 0; i < 2; ++i)
            #pragma unroll
            for (int j = 0; j < 2 * WB; ++j)
                acc[i][j] = __builtin_amdgcn_wmma_f32_16x16x32_f16(
                    false, af[i], false, bf[j], (short)0, acc[i][j], false, false);

        __syncthreads();   // reads of buf[cur] done; buf[cur^1] stores visible next iter
    }

    // ---- epilogue + store (C/D layout: lane -> N, VGPR r -> M = half*8 + r) ----
    #pragma unroll
    for (int i = 0; i < 2; ++i) {
        #pragma unroll
        for (int j = 0; j < 2 * WB; ++j) {
            #pragma unroll
            for (int r = 0; r < 8; ++r) {
                int m = bm0 + wm + i * 16 + half * 8 + r;
                int n = bn0 + wn + j * 16 + l15;
                if (m < M && n < Ncols) {
                    float x = acc[i][j][r];
                    float y;
                    if (mode == EPI_SIGMOID) {
                        y = 1.0f / (1.0f + __expf(-(x + bias0)));
                    } else if (mode == EPI_BIAS) {
                        y = x + aux[n];
                    } else {
                        y = x * scale;
                    }
                    C[coff + (long)m * c_rs + n] = y;
                }
            }
        }
    }
}

// ---------------------------------------------------------------------------
// Exclusive cumprod of (1-p) along S, per (n,t) row; one wave per row.
// In:  pc holds p (sigmoid output).  Out: pc <- p * cumprod,  icl <- 1/clamp.
// ---------------------------------------------------------------------------
__global__ __launch_bounds__(256)
void cumprod_kernel(float* __restrict__ pc, float* __restrict__ icl)
{
    long row = ((long)blockIdx.x * blockDim.x + threadIdx.x) >> 5;
    int lane = threadIdx.x & 31;
    if (row >= (long)NN * TT) return;
    float* prow = pc  + row * SS;
    float* irow = icl + row * SS;

    float carry = 1.0f;
    for (int c = 0; c < SS / 32; ++c) {
        float p  = prow[c * 32 + lane];
        float om = 1.0f - p;
        float incl = om;
        #pragma unroll
        for (int off = 1; off < 32; off <<= 1) {
            float t = __shfl_up(incl, off);
            if (lane >= off) incl *= t;
        }
        float ex = __shfl_up(incl, 1);
        if (lane == 0) ex = 1.0f;
        float cp = carry * ex;                       // exclusive cumprod
        float cl = fminf(fmaxf(cp, 1e-6f), 1.0f);
        prow[c * 32 + lane] = p * cp;                // p_choose * cumprod_1mp
        irow[c * 32 + lane] = 1.0f / cl;             // 1 / cumprod_clamp
        carry *= __shfl(incl, 31);
    }
}

// ---------------------------------------------------------------------------
// Sequential alpha recurrence over T (the serial critical path).
// One block per head-row n, 512 threads (thread s owns source position s).
//   alpha_t = clip(pc_t * cumsum_s(alpha_{t-1} * icl_t), 0, 1)
// Carries the UNFIXED alpha; stores the mass-preserving fixed row in place.
// Cross-wave scratch is parity-double-buffered: 2 barriers per step.
// ---------------------------------------------------------------------------
__global__ __launch_bounds__(512)
void alpha_scan_kernel(float* __restrict__ pc, const float* __restrict__ icl)
{
    const int n = blockIdx.x;
    const int s = threadIdx.x;
    const int lane = s & 31, wid = s >> 5;           // 16 waves
    __shared__ float wsum[2][16];
    __shared__ float wtot[2][16];

    float aprev = (s == 0) ? 1.0f : 0.0f;
    const long base = (long)n * TT * SS;

    for (int t = 0; t < TT; ++t) {
        const int pb = t & 1;
        const long row = base + (long)t * SS;
        float g = aprev * icl[row + s];

        // inclusive block-wide cumsum: wave shuffle scan + cross-wave prefix
        float incl = g;
        #pragma unroll
        for (int off = 1; off < 32; off <<= 1) {
            float tmp = __shfl_up(incl, off);
            if (lane >= off) incl += tmp;
        }
        if (lane == 31) wsum[pb][wid] = incl;
        __syncthreads();
        float prefix = 0.0f;
        for (int w = 0; w < wid; ++w) prefix += wsum[pb][w];
        float csum = incl + prefix;

        float a = fminf(fmaxf(pc[row + s] * csum, 0.0f), 1.0f);

        // mass preservation: alpha[S-1] = 1 - clip(sum(alpha[0..S-2]))
        float part = (s < SS - 1) ? a : 0.0f;
        #pragma unroll
        for (int off = 16; off > 0; off >>= 1) part += __shfl_down(part, off);
        if (lane == 0) wtot[pb][wid] = part;
        __syncthreads();
        float out = a;
        if (s == SS - 1) {
            float tot = 0.0f;
            #pragma unroll
            for (int w = 0; w < 16; ++w) tot += wtot[pb][w];
            out = 1.0f - fminf(fmaxf(tot, 0.0f), 1.0f);
        }
        pc[row + s] = out;      // fixed alpha stored for the attn GEMM
        aprev = a;              // UNFIXED alpha carried (matches reference scan)
    }
}

// ---------------------------------------------------------------------------
// Launch: q/k/v proj -> energy+sigmoid -> cumprod -> serial scan -> attn -> out
// ---------------------------------------------------------------------------
extern "C" void kernel_launch(void* const* d_in, const int* in_sizes, int n_in,
                              void* d_out, int out_size, void* d_ws, size_t ws_size,
                              hipStream_t stream)
{
    const float* q     = (const float*)d_in[0];
    const float* k     = (const float*)d_in[1];
    const float* v     = (const float*)d_in[2];
    const float* Wq    = (const float*)d_in[3];
    const float* Wk    = (const float*)d_in[4];
    const float* Wv    = (const float*)d_in[5];
    const float* Wo    = (const float*)d_in[6];
    const float* bo    = (const float*)d_in[7];
    const float* ebias = (const float*)d_in[8];
    float* out = (float*)d_out;

    float* ws    = (float*)d_ws;
    float* qp    = ws;                 // [T*B, E]      4M floats
    float* kp    = qp + QKV;           // [S*B, E]
    float* vp    = kp + QKV;           // [S*B, E]
    float* pc    = vp + QKV;           // [N,T,S]  p -> p*cp -> alpha (in place)
    float* icl   = pc + NTS;           // [N,T,S]  1/cumprod_clamp
    float* attnR = icl + NTS;          // [T,B,E] merged-head attn

    const dim3 blk(256);
    const float scaling = 0.125f;      // HEAD_DIM^-0.5

    // 1) projections: [4096,1024] = [4096,1024] @ [1024,1024], 128x128 tiles
    gemm_wmma_f16<2><<<dim3(8, 32, 1), blk, 0, stream>>>(
        q, EE, 0, 0, 1,  Wq, EE, 1, 0, 0, 1,  qp, EE, 0, 0, 1,
        TT * BB, EE, EE, EPI_SCALE, nullptr, scaling);
    gemm_wmma_f16<2><<<dim3(8, 32, 1), blk, 0, stream>>>(
        k, EE, 0, 0, 1,  Wk, EE, 1, 0, 0, 1,  kp, EE, 0, 0, 1,
        SS * BB, EE, EE, EPI_SCALE, nullptr, 1.0f);
    gemm_wmma_f16<2><<<dim3(8, 32, 1), blk, 0, stream>>>(
        v, EE, 0, 0, 1,  Wv, EE, 1, 0, 0, 1,  vp, EE, 0, 0, 1,
        SS * BB, EE, EE, EPI_SCALE, nullptr, 1.0f);

    // 2) energy + sigmoid: per head n, [T,S] = qp_n[T,D] @ kp_n[S,D]^T
    //    head base offset = (n/16)*1024 + (n%16)*64, row stride B*E = 8192
    gemm_wmma_f16<2><<<dim3(SS / 128, TT / 128, NN), blk, 0, stream>>>(
        qp, (long)BB * EE, EE, DD, HH,
        kp, 1, (long)BB * EE, EE, DD, HH,
        pc, SS, (long)TT * SS, 0, 1,
        TT, SS, DD, EPI_SIGMOID, ebias, 1.0f);

    // 3) exclusive cumprod along S (one wave per (n,t) row)
    cumprod_kernel<<<(NN * TT) / 8, blk, 0, stream>>>(pc, icl);

    // 4) serial alpha recurrence over T (one block per head)
    alpha_scan_kernel<<<NN, 512, 0, stream>>>(pc, icl);

    // 5) attn: per head, [T,D] = alpha[T,S] @ vp_n[S,D]; write merged [T,B,E]
    gemm_wmma_f16<1><<<dim3(DD / 64, TT / 128, NN), blk, 0, stream>>>(
        pc, SS, (long)TT * SS, 0, 1,
        vp, (long)BB * EE, 1, EE, DD, HH,
        attnR, (long)BB * EE, EE, DD, HH,
        TT, DD, SS, EPI_SCALE, nullptr, 1.0f);

    // 6) output projection + bias: [4096,1024] = attnR @ Wo + bo
    gemm_wmma_f16<2><<<dim3(8, 32, 1), blk, 0, stream>>>(
        attnR, EE, 0, 0, 1,  Wo, EE, 1, 0, 0, 1,  out, EE, 0, 0, 1,
        TT * BB, EE, EE, EPI_BIAS, bo, 1.0f);
}
